// DecoderScaledDotProductAttention_2671469658567
// MI455X (gfx1250) — matrix operand order repaired and checked by reading, not compiled, not run on hardware
//
#include <hip/hip_runtime.h>
#include <math.h>

typedef __attribute__((ext_vector_type(16))) _Float16 v16h;
typedef __attribute__((ext_vector_type(8)))  float    v8f;

#define S_LEN 2048
#define D_DIM 128
#define BM    64    // query rows per workgroup (4 waves x 16)
#define BN    32    // keys per iteration (one WMMA K=32 for PV)
#define NWAVE 4

union HU {                    // one WMMA 16-half operand
  v16h  v;
  _Float16 h[16];
  uint4 q[2];
};
union H4 {                    // 4 halfs <-> 8 bytes
  _Float16 h[4];
  uint2 u;
};

// 16-lane (DPP row) reductions: stay within lanes 0-15 / 16-31 half-groups,
// matching the WMMA C/D layout. Pure VALU -> co-executes with the WMMA pipe.
__device__ __forceinline__ float half_max16(float x) {
  x = fmaxf(x, __int_as_float(__builtin_amdgcn_update_dpp(
          0, __float_as_int(x), 0xB1, 0xF, 0xF, true)));   // quad_perm(1,0,3,2)
  x = fmaxf(x, __int_as_float(__builtin_amdgcn_update_dpp(
          0, __float_as_int(x), 0x4E, 0xF, 0xF, true)));   // quad_perm(2,3,0,1)
  x = fmaxf(x, __int_as_float(__builtin_amdgcn_update_dpp(
          0, __float_as_int(x), 0x141, 0xF, 0xF, true)));  // row_half_mirror
  x = fmaxf(x, __int_as_float(__builtin_amdgcn_update_dpp(
          0, __float_as_int(x), 0x140, 0xF, 0xF, true)));  // row_mirror
  return x;
}
__device__ __forceinline__ float half_sum16(float x) {
  x += __int_as_float(__builtin_amdgcn_update_dpp(
          0, __float_as_int(x), 0xB1, 0xF, 0xF, true));
  x += __int_as_float(__builtin_amdgcn_update_dpp(
          0, __float_as_int(x), 0x4E, 0xF, 0xF, true));
  x += __int_as_float(__builtin_amdgcn_update_dpp(
          0, __float_as_int(x), 0x141, 0xF, 0xF, true));
  x += __int_as_float(__builtin_amdgcn_update_dpp(
          0, __float_as_int(x), 0x140, 0xF, 0xF, true));
  return x;
}

__global__ __launch_bounds__(128)
void fa_causal_wmma_kernel(const float* __restrict__ qg,
                           const float* __restrict__ kg,
                           const float* __restrict__ vg,
                           float* __restrict__ og) {
  // LDS: K tile row-major [32][128], V tile transposed [128][32], per-wave P scratch [16][32]
  __shared__ __align__(16) _Float16 Klds[BN * D_DIM];
  __shared__ __align__(16) _Float16 Vt[D_DIM * BN];
  __shared__ __align__(16) _Float16 Plds[NWAVE * 16 * BN];

  const int tid  = threadIdx.x;
  const int wid  = tid >> 5;
  const int lane = tid & 31;
  const int l15  = lane & 15;
  const int hi   = lane >> 4;          // lane half selects K/M sub-range per ISA layouts

  const int qbase = blockIdx.x * BM;
  const int bh    = blockIdx.y;        // b*H + h

  const float scale = 0.08838834764831845f;   // 1/sqrt(128), folded into Q

  const float* qp = qg + (size_t)bh * S_LEN * D_DIM;
  const float* kp = kg + (size_t)bh * S_LEN * D_DIM;
  const float* vp = vg + (size_t)bh * S_LEN * D_DIM;
  float*       op = og + (size_t)bh * S_LEN * D_DIM;

  // ---- Load this wave's 16x128 Q tile into 4 A-operands (16x32 f16 each), pre-scaled.
  // A layout (16-bit, 16x32): lane m=l15; lanes<16: K = {c..c+7, c+16..c+23}; lanes>=16: +8.
  const int qrow = qbase + wid * 16 + l15;
  const float* qrp = qp + (size_t)qrow * D_DIM;
  v16h qa[4];
#pragma unroll
  for (int kc = 0; kc < 4; ++kc) {
    HU a;
    const int d0 = kc * 32 + hi * 8;
    float4 f0 = *(const float4*)(qrp + d0);
    float4 f1 = *(const float4*)(qrp + d0 + 4);
    a.h[0]=(_Float16)(f0.x*scale); a.h[1]=(_Float16)(f0.y*scale);
    a.h[2]=(_Float16)(f0.z*scale); a.h[3]=(_Float16)(f0.w*scale);
    a.h[4]=(_Float16)(f1.x*scale); a.h[5]=(_Float16)(f1.y*scale);
    a.h[6]=(_Float16)(f1.z*scale); a.h[7]=(_Float16)(f1.w*scale);
    const int d1 = kc * 32 + 16 + hi * 8;
    f0 = *(const float4*)(qrp + d1);
    f1 = *(const float4*)(qrp + d1 + 4);
    a.h[ 8]=(_Float16)(f0.x*scale); a.h[ 9]=(_Float16)(f0.y*scale);
    a.h[10]=(_Float16)(f0.z*scale); a.h[11]=(_Float16)(f0.w*scale);
    a.h[12]=(_Float16)(f1.x*scale); a.h[13]=(_Float16)(f1.y*scale);
    a.h[14]=(_Float16)(f1.z*scale); a.h[15]=(_Float16)(f1.w*scale);
    qa[kc] = a.v;
  }

  // ---- Online-softmax state. C/D layout: VGPR r, lanes<16 -> M=r, lanes>=16 -> M=r+8.
  float mrow[8], lrow[8];
  v8f acc[8];                 // 8 D-tiles of 16 cols, 16x16 f32 each
  v8f zero = {};
#pragma unroll
  for (int r = 0; r < 8; ++r) { mrow[r] = -INFINITY; lrow[r] = 0.0f; }
#pragma unroll
  for (int dt = 0; dt < 8; ++dt) acc[dt] = zero;

  const int wrow0    = qbase + wid * 16;   // this wave's min query row (uniform)
  const int wrow_max = wrow0 + 15;         // this wave's max query row (uniform)
  const int ntiles   = 2 * blockIdx.x + 2; // (qbase+BM)/BN

  for (int jt = 0; jt < ntiles; ++jt) {
    const int j = jt * BN;
    // scalar (SGPR) per-iteration bases; all vector offsets below are loop-invariant
    const float* kpj = kp + (size_t)j * D_DIM;
    const float* vpj = vp + (size_t)j * D_DIM;

    __syncthreads();                     // previous iteration's LDS reads done
    // ---- Stage K rows j..j+31 -> Klds (row-major f16). Whole wave reads 512B
    //      contiguous per load; uint2 LDS stores are conflict-free.
#pragma unroll
    for (int i = 0; i < 8; ++i) {
      const int off = (wid + 4 * i) * D_DIM + lane * 4;  // invariant vector offset
      float4 fk = *(const float4*)(kpj + off);
      H4 hk;
      hk.h[0]=(_Float16)fk.x; hk.h[1]=(_Float16)fk.y;
      hk.h[2]=(_Float16)fk.z; hk.h[3]=(_Float16)fk.w;
      *(uint2*)(&Klds[off]) = hk.u;
    }
    // ---- Stage V rows j..j+31 -> Vt (transposed, d-major f16). row==lane so the
    //      b16 scatter spreads across banks.
#pragma unroll
    for (int i = 0; i < 8; ++i) {
      const int g = wid + 4 * i;
      float4 fv = *(const float4*)(vpj + lane * D_DIM + g * 4);
      const int d = g * 4;
      Vt[(d + 0) * BN + lane] = (_Float16)fv.x;
      Vt[(d + 1) * BN + lane] = (_Float16)fv.y;
      Vt[(d + 2) * BN + lane] = (_Float16)fv.z;
      Vt[(d + 3) * BN + lane] = (_Float16)fv.w;
    }
    if (jt + 1 < ntiles) {               // cover the full 16KB next K and V tiles
      const float* kpn = kpj + BN * D_DIM;
      const float* vpn = vpj + BN * D_DIM;
      __builtin_prefetch(kpn + tid * 32, 0, 1);
      __builtin_prefetch(kpn + tid * 32 + 16, 0, 1);
      __builtin_prefetch(vpn + tid * 32, 0, 1);
      __builtin_prefetch(vpn + tid * 32 + 16, 0, 1);
    }
    __syncthreads();

    // Tiles entirely above this wave's rows are fully masked -> skip (uniform branch,
    // EXEC stays all-ones; no barriers inside).
    if (j > wrow_max) continue;

    // ---- S = (Q*scale) @ K^T : two 16x16 f32 N-tiles, K-dim 128 in 4 chunks.
    // B layout (16-bit, 32x16): lane n=l15; lanes<16 hold K {0..7, 16..23}, lanes>=16 hold K {8..15, 24..31}.
    HU kb[8];
#pragma unroll
    for (int kc = 0; kc < 4; ++kc) {
      const int koff = kc * 32 + hi * 8;
      const _Float16* kr0 = &Klds[l15 * D_DIM + koff];
      kb[kc].q[0] = *(const uint4*)(kr0);
      kb[kc].q[1] = *(const uint4*)(kr0 + 16);
      const _Float16* kr1 = &Klds[(16 + l15) * D_DIM + koff];
      kb[4 + kc].q[0] = *(const uint4*)(kr1);
      kb[4 + kc].q[1] = *(const uint4*)(kr1 + 16);
    }
    v8f c0 = zero, c1 = zero;
#pragma unroll
    for (int kc = 0; kc < 4; ++kc) {
      c0 = __builtin_amdgcn_wmma_f32_16x16x32_f16(false, qa[kc], false, kb[kc].v,
                                                  (short)0, c0, false, false);
      c1 = __builtin_amdgcn_wmma_f32_16x16x32_f16(false, qa[kc], false, kb[4 + kc].v,
                                                  (short)0, c1, false, false);
    }

    // ---- Causal mask + online softmax, specialized on a wave-uniform "crosses
    // the diagonal" flag. DPP16 reductions, no LDS round-trips.
    _Float16* Pw = &Plds[wid * 16 * BN];
    auto softmax_block = [&](bool masked) {
#pragma unroll
      for (int r = 0; r < 8; ++r) {
        const int mg = wrow0 + r + 8 * hi;            // global query row
        float s0 = c0[r];
        float s1 = c1[r];
        if (masked) {
          s0 = (j + l15      <= mg) ? s0 : -INFINITY;
          s1 = (j + 16 + l15 <= mg) ? s1 : -INFINITY;
        }
        const float mt   = half_max16(fmaxf(s0, s1));
        const float mnew = fmaxf(mrow[r], mt);
        const float corr = __expf(mrow[r] - mnew);
        const float p0 = __expf(s0 - mnew);
        const float p1 = __expf(s1 - mnew);
        const float ps = half_sum16(p0 + p1);
        lrow[r] = lrow[r] * corr + ps;
        mrow[r] = mnew;
#pragma unroll
        for (int dt = 0; dt < 8; ++dt) acc[dt][r] *= corr;
        // C-layout -> row-major P scratch (per-wave private; same-wave LDS ops are
        // processed in order, so no barrier needed before reading it back).
        Pw[(r + 8 * hi) * BN + l15]      = (_Float16)p0;
        Pw[(r + 8 * hi) * BN + 16 + l15] = (_Float16)p1;
      }
    };
    if (j + BN - 1 > wrow0) softmax_block(true);
    else                    softmax_block(false);

    // ---- O += P @ V : A = P (16x32 f16 A-layout), B = V columns from Vt.
    HU pa;
    const _Float16* pr = &Pw[l15 * BN];
    pa.q[0] = *(const uint4*)(pr + hi * 8);
    pa.q[1] = *(const uint4*)(pr + 16 + hi * 8);
    HU vb[8];
#pragma unroll
    for (int dt = 0; dt < 8; ++dt) {
      const _Float16* vr = &Vt[(dt * 16 + l15) * BN + hi * 8];
      vb[dt].q[0] = *(const uint4*)(vr);
      vb[dt].q[1] = *(const uint4*)(vr + 16);
    }
#pragma unroll
    for (int dt = 0; dt < 8; ++dt)
      acc[dt] = __builtin_amdgcn_wmma_f32_16x16x32_f16(false, pa.v, false, vb[dt].v,
                                                       (short)0, acc[dt], false, false);
  }

  // ---- Epilogue: normalize by row sum and store fp32.
#pragma unroll
  for (int r = 0; r < 8; ++r) {
    const float inv = 1.0f / lrow[r];
    const int row = wrow0 + r + 8 * hi;
    float* orow = op + (size_t)row * D_DIM;
#pragma unroll
    for (int dt = 0; dt < 8; ++dt)
      orow[dt * 16 + l15] = acc[dt][r] * inv;
  }
}

extern "C" void kernel_launch(void* const* d_in, const int* in_sizes, int n_in,
                              void* d_out, int out_size, void* d_ws, size_t ws_size,
                              hipStream_t stream) {
  (void)in_sizes; (void)n_in; (void)out_size; (void)d_ws; (void)ws_size;
  const float* q = (const float*)d_in[0];
  const float* k = (const float*)d_in[1];
  const float* v = (const float*)d_in[2];
  // d_in[3] is the causal mask; causality is applied analytically in-kernel.
  float* out = (float*)d_out;
  dim3 grid(S_LEN / BM, 2 * 16);   // (q-tiles, B*H) = (32, 32)
  dim3 block(128);                 // 4 wave32
  hipLaunchKernelGGL(fa_causal_wmma_kernel, grid, block, 0, stream, q, k, v, out);
}
